// YuanExperts_5317169512809
// MI455X (gfx1250) — compile-verified, matrix operand order checked by reading, not run on hardware
//
#include <hip/hip_runtime.h>
#include <math.h>

#define T_TOK 2048
#define H_DIM 1024
#define E_NUM 32
#define I_DIM 2048

typedef __attribute__((ext_vector_type(16))) __bf16 v16bf;
typedef __attribute__((ext_vector_type(8)))  __bf16 v8bf;
typedef __attribute__((ext_vector_type(8)))  float  v8f;

union FragA { v16bf v; v8bf h[2]; };

static __device__ __forceinline__ unsigned short f2bf(float f) {
  unsigned u = __float_as_uint(f);
  u += 0x7FFFu + ((u >> 16) & 1u);           // round-to-nearest-even
  return (unsigned short)(u >> 16);
}

static __device__ __forceinline__ void store4bf(unsigned short* dst, float4 f) {
  unsigned a = (unsigned)f2bf(f.x) | ((unsigned)f2bf(f.y) << 16);
  unsigned b = (unsigned)f2bf(f.z) | ((unsigned)f2bf(f.w) << 16);
  *reinterpret_cast<uint2*>(dst) = make_uint2(a, b);
}

static __device__ __forceinline__ float fast_rcp(float x) {
  return __builtin_amdgcn_rcpf(x);           // raw v_rcp_f32
}

// ---------------------------------------------------------------- zero
__global__ void k_zero(float* out, int n, int* counts, int* fill) {
  int i = blockIdx.x * blockDim.x + threadIdx.x;
  if (i < n) out[i] = 0.f;
  if (i < E_NUM) { counts[i] = 0; fill[i] = 0; }
}

// ---------------------------------------------------------------- router
// mix = x @ wqkv.T ; logits[e] = sum_f softmax_f(q[e]*k[f]) * v[f]
__global__ void k_router(const float* __restrict__ x, const float* __restrict__ wqkv,
                         float* __restrict__ logits) {
  __shared__ float mix[96];
  const int t = blockIdx.x;
  const int i = threadIdx.x;                 // 0..95
  const float4* xr = reinterpret_cast<const float4*>(x + (size_t)t * H_DIM);
  const float4* wr = reinterpret_cast<const float4*>(wqkv + (size_t)i * H_DIM);
  float acc = 0.f;
  for (int k = 0; k < H_DIM / 4; ++k) {
    float4 a = xr[k], b = wr[k];
    acc += a.x * b.x + a.y * b.y + a.z * b.z + a.w * b.w;
  }
  mix[i] = acc;
  __syncthreads();
  if (i < E_NUM) {
    float q = mix[i];
    float m = -INFINITY;
    for (int f = 0; f < E_NUM; ++f) m = fmaxf(m, q * mix[32 + f]);
    float s = 0.f, num = 0.f;
    for (int f = 0; f < E_NUM; ++f) {
      float p = __expf(q * mix[32 + f] - m);
      s += p;
      num += p * mix[64 + f];
    }
    logits[(size_t)t * E_NUM + i] = num * fast_rcp(s);
  }
}

// ---------------------------------------------------------------- top-2
__global__ void k_topk(const float* __restrict__ logits, int* __restrict__ topi,
                       float* __restrict__ topw, int* __restrict__ counts) {
  int t = blockIdx.x * blockDim.x + threadIdx.x;
  if (t >= T_TOK) return;
  const float* l = logits + (size_t)t * E_NUM;
  float v0 = -INFINITY, v1 = -INFINITY; int i0 = 0, i1 = 0;
  for (int e = 0; e < E_NUM; ++e) {
    float v = l[e];
    if (v > v0) { v1 = v0; i1 = i0; v0 = v; i0 = e; }
    else if (v > v1) { v1 = v; i1 = e; }
  }
  float p = __expf(v1 - v0);                 // softmax over the selected pair
  float inv = fast_rcp(1.f + p);
  topi[2 * t] = i0; topi[2 * t + 1] = i1;
  topw[2 * t] = inv; topw[2 * t + 1] = p * inv;
  atomicAdd(&counts[i0], 1);
  atomicAdd(&counts[i1], 1);
}

__global__ void k_scan(const int* __restrict__ counts, int* __restrict__ offsets) {
  if (blockIdx.x == 0 && threadIdx.x == 0) {
    int acc = 0;
    for (int e = 0; e < E_NUM; ++e) { offsets[e] = acc; acc += counts[e]; }
  }
}

__global__ void k_assign(const int* __restrict__ topi, const float* __restrict__ topw,
                         const int* __restrict__ offsets, int* __restrict__ fill,
                         int* __restrict__ token_list, float* __restrict__ token_w) {
  int t = blockIdx.x * blockDim.x + threadIdx.x;
  if (t >= T_TOK) return;
  for (int j = 0; j < 2; ++j) {
    int e = topi[2 * t + j];
    float w = topw[2 * t + j];
    int pos = atomicAdd(&fill[e], 1);
    token_list[offsets[e] + pos] = t;
    token_w[offsets[e] + pos] = w;
  }
}

// ---------------------------------------------------------------- GEMM1 + SwiGLU
// act[row, j] = silu(Xg . w13[j]) * (Xg . w13[I+j]), bf16 out.
// Tile: 32 tokens x 64 act-cols per block; 8 waves, each wave = one 16x16
// gate tile + one 16x16 up tile; K=H in steps of 32.
// Grid: blockIdx.x = m-tile (fastest) so consecutive blocks reuse the same
// weight tile from L2; blockIdx.y = n-tile; blockIdx.z = expert.
__global__ __launch_bounds__(256)
void k_gemm1(const float* __restrict__ x, const float* __restrict__ w13,
             const int* __restrict__ counts, const int* __restrict__ offsets,
             const int* __restrict__ token_list, unsigned short* __restrict__ act) {
  __shared__ __attribute__((aligned(16))) unsigned short lA[32 * 32];
  __shared__ __attribute__((aligned(16))) unsigned short lB[128 * 32];
  __shared__ __attribute__((aligned(16))) unsigned short lC[32 * 64];

  const int e = blockIdx.z;
  const int cnt = counts[e];
  const int mbase = blockIdx.x * 32;
  if (mbase >= cnt) return;
  const int off = offsets[e];
  const int nbase = blockIdx.y * 64;

  const int tid = threadIdx.x;
  const int lane = tid & 31, wid = tid >> 5;
  const int wm = wid & 1, wn = wid >> 1;
  const int half = lane >> 4, l = lane & 15;

  const int arow = tid >> 3, akq = tid & 7;
  const int tokIdx = (mbase + arow < cnt) ? (mbase + arow) : mbase;
  const int tok = token_list[off + tokIdx];
  const float4* xrow = reinterpret_cast<const float4*>(x + (size_t)tok * H_DIM);
  const float* w13e = w13 + (size_t)e * (2 * (size_t)I_DIM) * H_DIM;

  v8f cg = {0.f, 0.f, 0.f, 0.f, 0.f, 0.f, 0.f, 0.f};
  v8f cu = cg;

  for (int k0 = 0; k0 < H_DIM; k0 += 32) {
    __syncthreads();
    // A: 32 rows x 32 k, fp32 -> bf16 (one float4 per thread)
    store4bf(&lA[arow * 32 + akq * 4], xrow[(k0 >> 2) + akq]);
    // B: gate rows [0,64) + up rows [64,128), fp32 -> bf16
    #pragma unroll
    for (int i = 0; i < 4; ++i) {
      int idx = tid + i * 256;
      int brow = idx >> 3, kq = idx & 7;
      int grow = (brow < 64) ? (nbase + brow) : (I_DIM + nbase + (brow - 64));
      float4 f = *reinterpret_cast<const float4*>(w13e + (size_t)grow * H_DIM + k0 + kq * 4);
      store4bf(&lB[brow * 32 + kq * 4], f);
    }
    __syncthreads();

    FragA fa, fbg, fbu;
    const int ar = wm * 16 + l;
    const int klo = half ? 8 : 0, khi = half ? 24 : 16;
    fa.h[0]  = *reinterpret_cast<const v8bf*>(&lA[ar * 32 + klo]);
    fa.h[1]  = *reinterpret_cast<const v8bf*>(&lA[ar * 32 + khi]);
    const int bg = wn * 16 + l;
    fbg.h[0] = *reinterpret_cast<const v8bf*>(&lB[bg * 32 + klo]);
    fbg.h[1] = *reinterpret_cast<const v8bf*>(&lB[bg * 32 + khi]);
    const int bu = 64 + wn * 16 + l;
    fbu.h[0] = *reinterpret_cast<const v8bf*>(&lB[bu * 32 + klo]);
    fbu.h[1] = *reinterpret_cast<const v8bf*>(&lB[bu * 32 + khi]);

    cg = __builtin_amdgcn_wmma_f32_16x16x32_bf16(false, fa.v, false, fbg.v, (short)0, cg, false, false);
    cu = __builtin_amdgcn_wmma_f32_16x16x32_bf16(false, fa.v, false, fbu.v, (short)0, cu, false, false);
  }

  // fused SwiGLU (fast sigmoid via v_rcp_f32), stage 32x64 bf16 tile in LDS
  #pragma unroll
  for (int r = 0; r < 8; ++r) {
    float g = cg[r], u = cu[r];
    float a = g * fast_rcp(1.f + __expf(-g)) * u;
    lC[(wm * 16 + half * 8 + r) * 64 + wn * 16 + l] = f2bf(a);
  }
  __syncthreads();
  // coalesced b128 store of the tile
  {
    int row = tid >> 3, cq = tid & 7;
    if (mbase + row < cnt) {
      uint4 v = *reinterpret_cast<const uint4*>(&lC[row * 64 + cq * 8]);
      *reinterpret_cast<uint4*>(&act[(size_t)(off + mbase + row) * I_DIM + nbase + cq * 8]) = v;
    }
  }
}

// ---------------------------------------------------------------- GEMM2 + scatter
// out[t, h] += w_t * sum_k act[row,k] * w2s[e][h,k]
// Tile: 32 tokens x 128 h-cols; each wave = two 16x16 tiles; K=I in steps of 32.
// A-tile (already bf16) is staged with the CDNA5 async global->LDS path.
// Grid: blockIdx.x = m-tile (fastest, L2 weight reuse), y = h-tile, z = expert.
__global__ __launch_bounds__(256)
void k_gemm2(const unsigned short* __restrict__ act, const float* __restrict__ w2,
             const int* __restrict__ counts, const int* __restrict__ offsets,
             const int* __restrict__ token_list, const float* __restrict__ token_w,
             float* __restrict__ out) {
  __shared__ __attribute__((aligned(16))) unsigned short lA[32 * 32];
  __shared__ __attribute__((aligned(16))) unsigned short lB[128 * 32];

  const int e = blockIdx.z;
  const int cnt = counts[e];
  const int mbase = blockIdx.x * 32;
  if (mbase >= cnt) return;
  const int off = offsets[e];
  const int hbase = blockIdx.y * 128;

  const int tid = threadIdx.x;
  const int lane = tid & 31, wid = tid >> 5;
  const int wm = wid & 1, wn = wid >> 1;
  const int half = lane >> 4, l = lane & 15;

  const int arow = tid >> 3, akq = tid & 7;
  const int mrowIdx = (mbase + arow < cnt) ? (mbase + arow) : mbase;
  const unsigned short* actRow = act + (size_t)(off + mrowIdx) * I_DIM;
  const float* w2e = w2 + (size_t)e * H_DIM * I_DIM;

  // per-thread LDS byte offset for the async A copy (low 32 bits of the
  // generic pointer to a __shared__ object == LDS offset)
  const unsigned ldsA = (unsigned)(uintptr_t)(&lA[arow * 32 + akq * 4]);

  v8f c0 = {0.f, 0.f, 0.f, 0.f, 0.f, 0.f, 0.f, 0.f};
  v8f c1 = c0;

  for (int k0 = 0; k0 < I_DIM; k0 += 32) {
    __syncthreads();
    // A: async DMA global->LDS, 8 bytes/lane (bf16 already), ASYNCcnt-tracked
    {
      unsigned long long gaddr = (unsigned long long)(uintptr_t)(actRow + k0 + akq * 4);
      asm volatile("global_load_async_to_lds_b64 %0, %1, off"
                   :: "v"(ldsA), "v"(gaddr) : "memory");
    }
    // B: 128 h-rows x 32 k fp32 -> bf16 (overlaps with the async A copy)
    #pragma unroll
    for (int i = 0; i < 4; ++i) {
      int idx = tid + i * 256;
      int brow = idx >> 3, kq = idx & 7;
      float4 f = *reinterpret_cast<const float4*>(w2e + (size_t)(hbase + brow) * I_DIM + k0 + kq * 4);
      store4bf(&lB[brow * 32 + kq * 4], f);
    }
    asm volatile("s_wait_asynccnt 0" ::: "memory");
    __syncthreads();

    FragA fa, fb0, fb1;
    const int ar = wm * 16 + l;
    const int klo = half ? 8 : 0, khi = half ? 24 : 16;
    fa.h[0]  = *reinterpret_cast<const v8bf*>(&lA[ar * 32 + klo]);
    fa.h[1]  = *reinterpret_cast<const v8bf*>(&lA[ar * 32 + khi]);
    const int b0 = wn * 32 + l;
    fb0.h[0] = *reinterpret_cast<const v8bf*>(&lB[b0 * 32 + klo]);
    fb0.h[1] = *reinterpret_cast<const v8bf*>(&lB[b0 * 32 + khi]);
    const int b1 = wn * 32 + 16 + l;
    fb1.h[0] = *reinterpret_cast<const v8bf*>(&lB[b1 * 32 + klo]);
    fb1.h[1] = *reinterpret_cast<const v8bf*>(&lB[b1 * 32 + khi]);

    c0 = __builtin_amdgcn_wmma_f32_16x16x32_bf16(false, fa.v, false, fb0.v, (short)0, c0, false, false);
    c1 = __builtin_amdgcn_wmma_f32_16x16x32_bf16(false, fa.v, false, fb1.v, (short)0, c1, false, false);
  }

  #pragma unroll
  for (int r = 0; r < 8; ++r) {
    int rl = mbase + wm * 16 + half * 8 + r;
    if (rl < cnt) {
      int t = token_list[off + rl];
      float w = token_w[off + rl];
      int col0 = hbase + wn * 32 + l;
      atomicAdd(&out[(size_t)t * H_DIM + col0], w * c0[r]);
      atomicAdd(&out[(size_t)t * H_DIM + col0 + 16], w * c1[r]);
    }
  }
}

// ---------------------------------------------------------------- launch
extern "C" void kernel_launch(void* const* d_in, const int* in_sizes, int n_in,
                              void* d_out, int out_size, void* d_ws, size_t ws_size,
                              hipStream_t stream) {
  (void)in_sizes; (void)n_in; (void)out_size; (void)ws_size;
  const float* x    = (const float*)d_in[0];
  const float* wqkv = (const float*)d_in[1];
  const float* w13  = (const float*)d_in[2];
  const float* w2   = (const float*)d_in[3];
  float* out = (float*)d_out;

  char* ws = (char*)d_ws;
  float* logits     = (float*)(ws + 0);        // 262144 B
  int*   topi       = (int*)  (ws + 262144);   // 16384 B
  float* topw       = (float*)(ws + 278528);   // 16384 B
  int*   counts     = (int*)  (ws + 294912);   // 128 B
  int*   offsets    = (int*)  (ws + 295040);   // 128 B
  int*   fill       = (int*)  (ws + 295168);   // 128 B
  int*   token_list = (int*)  (ws + 295296);   // 16384 B
  float* token_w    = (float*)(ws + 311680);   // 16384 B
  unsigned short* act = (unsigned short*)(ws + 331776); // 4096*2048 bf16 = 16 MB

  const int nOut = T_TOK * H_DIM;
  k_zero<<<(nOut + 255) / 256, 256, 0, stream>>>(out, nOut, counts, fill);
  k_router<<<T_TOK, 96, 0, stream>>>(x, wqkv, logits);
  k_topk<<<T_TOK / 256, 256, 0, stream>>>(logits, topi, topw, counts);
  k_scan<<<1, 32, 0, stream>>>(counts, offsets);
  k_assign<<<T_TOK / 256, 256, 0, stream>>>(topi, topw, offsets, fill, token_list, token_w);

  // m-tile fastest: consecutive blocks share the same weight tile (L2 reuse)
  dim3 g1((T_TOK + 31) / 32, I_DIM / 64, E_NUM);
  k_gemm1<<<g1, 256, 0, stream>>>(x, w13, counts, offsets, token_list, act);

  dim3 g2((T_TOK + 31) / 32, H_DIM / 128, E_NUM);
  k_gemm2<<<g2, 256, 0, stream>>>(act, w2, counts, offsets, token_list, token_w, out);
}
